// GNNMOT_74131135529728
// MI455X (gfx1250) — matrix operand order repaired and checked by reading, not compiled, not run on hardware
//
#include <hip/hip_runtime.h>
#include <hip/hip_bf16.h>
#include <math.h>

typedef __attribute__((ext_vector_type(16))) _Float16 v16h;
typedef __attribute__((ext_vector_type(8)))  _Float16 v8h;
typedef __attribute__((ext_vector_type(4)))  _Float16 v4h;
typedef __attribute__((ext_vector_type(8)))  float    v8f;

// ---------------------------------------------------------------------------
// WMMA helpers (wave32, 16x16x32 f16 -> f32).  All GEMMs computed transposed:
//   D[16n x 16m] = A(W)[16n x 32k] * B(act.T)[32k x 16m] + C
// so A = weights (pre-swizzled LDS fragments), B = activations read straight
// from row-major [m][k] f16 tiles, D stores 8 contiguous features per lane.
// ---------------------------------------------------------------------------
__device__ __forceinline__ v8f wmma16(v16h a, v16h b, v8f c) {
  return __builtin_amdgcn_wmma_f32_16x16x32_f16(
      false, a, false, b, (short)0, c, false, false);
}

// Pre-swizzled fragment: 16 contiguous halfs per lane -> 2x ds_load_b128.
__device__ __forceinline__ v16h load_frag_sw(const _Float16* frag, int lane) {
  const v8h* q = (const v8h*)(frag + (lane << 4));
  const v8h lo = q[0], hi = q[1];
  return __builtin_shufflevector(lo, hi, 0, 1, 2, 3, 4, 5, 6, 7,
                                 8, 9, 10, 11, 12, 13, 14, 15);
}

// B fragment of act.T, act row-major [16][ldk] f16.
// lane: col(=act row) = lane&15, k = kt*32 + (lane>>4)*16 + 0..15 (contiguous).
__device__ __forceinline__ v16h load_bT(const _Float16* act, int ldk, int kt, int lane) {
  const v8h* q = (const v8h*)(act + (lane & 15) * ldk + kt * 32 + ((lane >> 4) << 4));
  return __builtin_shufflevector(q[0], q[1], 0, 1, 2, 3, 4, 5, 6, 7,
                                 8, 9, 10, 11, 12, 13, 14, 15);
}

// Stage weight W (row-major [NA][K] f32) into swizzled A fragments (f16).
// Fragment t = nt*KT + kt holds rows n = nt*16+(lane&15),
// k = kt*32 + (lane>>4)*16 + 0..15; element order matches A layout.
template <int K, int NA>
__device__ __forceinline__ void stage_a_sw(_Float16* __restrict__ dst,
                                           const float* __restrict__ W,
                                           int tid, int nth) {
  constexpr int KT = K / 32;
  for (int i = tid * 4; i < K * NA; i += nth * 4) {
    const int t = i >> 9, rem = i & 511;
    const int lane = rem >> 4, e = rem & 15;
    const int nt = t / KT, kt = t - nt * KT;
    const int n = nt * 16 + (lane & 15);
    const int k = kt * 32 + ((lane >> 4) << 4) + e;
    const float4 w = *(const float4*)(W + n * K + k);
    v4h o; o[0] = (_Float16)w.x; o[1] = (_Float16)w.y;
           o[2] = (_Float16)w.z; o[3] = (_Float16)w.w;
    *(v4h*)(dst + i) = o;
  }
}

// C init from per-row (feature) bias: c[r] = b[nt*16 + (lane>>4)*8 + r].
__device__ __forceinline__ v8f bias_rows(const float* b, int nt, int lane) {
  const float* p = b + nt * 16 + ((lane >> 4) << 3);
  const float4 x = *(const float4*)p;
  const float4 y = *(const float4*)(p + 4);
  v8f c; c[0] = x.x; c[1] = x.y; c[2] = x.z; c[3] = x.w;
         c[4] = y.x; c[5] = y.y; c[6] = y.z; c[7] = y.w;
  return c;
}

// D store (transposed) -> f16 LDS tile dstT[m][n], relu. One b128 per call.
__device__ __forceinline__ void store_dT_relu_f16(_Float16* dstT, int ldn, int nt,
                                                  int lane, v8f c) {
  v8h o;
  #pragma unroll
  for (int r = 0; r < 8; ++r) o[r] = (_Float16)fmaxf(c[r], 0.f);
  *(v8h*)(dstT + (lane & 15) * ldn + nt * 16 + ((lane >> 4) << 3)) = o;
}

// D store (transposed) -> f32 global out[m][n]. Two b128 per call.
__device__ __forceinline__ void store_dT_f32(float* out, int ldn, int nt,
                                             int lane, v8f c) {
  float* p = out + (size_t)(lane & 15) * ldn + nt * 16 + ((lane >> 4) << 3);
  *(float4*)p       = make_float4(c[0], c[1], c[2], c[3]);
  *(float4*)(p + 4) = make_float4(c[4], c[5], c[6], c[7]);
}

__device__ __forceinline__ float sigmoidf_(float x) { return 1.f / (1.f + __expf(-x)); }

// ---------------------------------------------------------------------------
// PointNet: per object (512 points, 5 -> 64 -> 128 -> 64), max over points.
// grid = 512 objects, block = 128 (4 waves). Writes h[obj][0:64].
// ---------------------------------------------------------------------------
__global__ __launch_bounds__(128) void pointnet_kernel(
    const float* __restrict__ det_pts, const float* __restrict__ trk_pts,
    const float* __restrict__ w1, const float* __restrict__ b1,
    const float* __restrict__ w2, const float* __restrict__ b2,
    const float* __restrict__ w3, const float* __restrict__ b3,
    float* __restrict__ hmat) {
  __shared__ alignas(32) _Float16 sW2A[16 * 512];   // W2: 8 nt x 2 kt frags
  __shared__ alignas(32) _Float16 sW3A[16 * 512];   // W3: 4 nt x 4 kt frags
  __shared__ alignas(16) float sW1[64 * 5];
  __shared__ alignas(32) float sB1[64], sB2[128], sB3[64];
  __shared__ alignas(32) _Float16 sH1[4][16 * 64];   // [p][k]
  __shared__ alignas(32) _Float16 sH2T[4][16 * 128]; // [p][n]
  __shared__ alignas(16) float sX[4][80];
  __shared__ float sMax[4][64];

  const int tid = threadIdx.x, wave = tid >> 5, lane = tid & 31;
  const int obj = blockIdx.x;
  const float* pts = (obj < 256) ? (det_pts + (size_t)obj * 512 * 5)
                                 : (trk_pts + (size_t)(obj - 256) * 512 * 5);

  stage_a_sw<64, 128>(sW2A, w2, tid, 128);
  stage_a_sw<128, 64>(sW3A, w3, tid, 128);
  for (int i = tid; i < 64 * 5; i += 128) sW1[i] = w1[i];
  if (tid < 64)  { sB1[tid] = b1[tid]; sB3[tid] = b3[tid]; }
  if (tid < 128) sB2[tid] = b2[tid];
  __syncthreads();

  float vmax[4][8];
  #pragma unroll
  for (int nt = 0; nt < 4; ++nt)
    #pragma unroll
    for (int r = 0; r < 8; ++r) vmax[nt][r] = -__builtin_inff();

  _Float16* h1  = sH1[wave];
  _Float16* h2t = sH2T[wave];
  float*    xt  = sX[wave];

  for (int mt = wave * 8; mt < wave * 8 + 8; ++mt) {
    const int p0 = mt * 16;
    for (int i = lane; i < 20; i += 32)
      *(float4*)(xt + i * 4) = *(const float4*)(pts + p0 * 5 + i * 4);
    // layer1 (K=5, VALU) -> relu -> [p][f] f16
    for (int i = lane * 4; i < 16 * 64; i += 128) {
      const int p = i >> 6, f0 = i & 63;
      v4h o;
      #pragma unroll
      for (int q = 0; q < 4; ++q) {
        const int f = f0 + q;
        float acc = sB1[f];
        #pragma unroll
        for (int k = 0; k < 5; ++k) acc += xt[p * 5 + k] * sW1[f * 5 + k];
        o[q] = (_Float16)fmaxf(acc, 0.f);
      }
      *(v4h*)(h1 + i) = o;
    }
    // layer2: H2.T = W2 @ H1.T  (relu, f16 LDS)
    {
      v16h bf0 = load_bT(h1, 64, 0, lane);
      v16h bf1 = load_bT(h1, 64, 1, lane);
      #pragma unroll
      for (int nt = 0; nt < 8; ++nt) {
        v8f c = bias_rows(sB2, nt, lane);
        c = wmma16(load_frag_sw(sW2A + (nt * 2 + 0) * 512, lane), bf0, c);
        c = wmma16(load_frag_sw(sW2A + (nt * 2 + 1) * 512, lane), bf1, c);
        store_dT_relu_f16(h2t, 128, nt, lane, c);
      }
    }
    // layer3: W3 @ H2.T (no relu), per-lane running max over this lane's point
    {
      v16h bf[4];
      #pragma unroll
      for (int ks = 0; ks < 4; ++ks) bf[ks] = load_bT(h2t, 128, ks, lane);
      #pragma unroll
      for (int nt = 0; nt < 4; ++nt) {
        v8f c = bias_rows(sB3, nt, lane);
        #pragma unroll
        for (int ks = 0; ks < 4; ++ks)
          c = wmma16(load_frag_sw(sW3A + (nt * 4 + ks) * 512, lane), bf[ks], c);
        #pragma unroll
        for (int r = 0; r < 8; ++r) vmax[nt][r] = fmaxf(vmax[nt][r], c[r]);
      }
    }
  }

  // reduce over the 16 point-lanes of each half, then across waves
  #pragma unroll
  for (int nt = 0; nt < 4; ++nt)
    #pragma unroll
    for (int r = 0; r < 8; ++r) {
      float v = vmax[nt][r];
      v = fmaxf(v, __shfl_xor(v, 1));
      v = fmaxf(v, __shfl_xor(v, 2));
      v = fmaxf(v, __shfl_xor(v, 4));
      v = fmaxf(v, __shfl_xor(v, 8));
      if ((lane & 15) == 0) sMax[wave][nt * 16 + ((lane >> 4) << 3) + r] = v;
    }
  __syncthreads();
  if (wave == 0) {
    for (int f = lane; f < 64; f += 32) {
      float m = sMax[0][f];
      #pragma unroll
      for (int w = 1; w < 4; ++w) m = fmaxf(m, sMax[w][f]);
      hmat[(size_t)obj * 128 + f] = m;
    }
  }
}

// ---------------------------------------------------------------------------
// Detection motion MLP: 9 -> 32 -> 64. grid = 256 rows, block = 64.
// ---------------------------------------------------------------------------
__global__ __launch_bounds__(64) void detmot_kernel(
    const float* __restrict__ boxes,
    const float* __restrict__ w1, const float* __restrict__ b1,
    const float* __restrict__ w2, const float* __restrict__ b2,
    float* __restrict__ hmat) {
  __shared__ float hid[32];
  const int i = blockIdx.x, t = threadIdx.x;
  if (t < 32) {
    float acc = b1[t];
    #pragma unroll
    for (int k = 0; k < 9; ++k) acc += boxes[i * 9 + k] * w1[t * 9 + k];
    hid[t] = fmaxf(acc, 0.f);
  }
  __syncthreads();
  float acc = b2[t];
  #pragma unroll
  for (int k = 0; k < 32; ++k) acc += hid[k] * w2[t * 32 + k];
  hmat[(size_t)i * 128 + 64 + t] = acc;
}

// ---------------------------------------------------------------------------
// LSTM layer0 input projection (K=9, VALU): xg[b*T+t][g].
// ---------------------------------------------------------------------------
__global__ __launch_bounds__(256) void xgemm0_kernel(
    const float* __restrict__ tb, const float* __restrict__ wih,
    float* __restrict__ xg) {
  const int row = blockIdx.x, g = threadIdx.x;
  float acc = 0.f;
  #pragma unroll
  for (int k = 0; k < 9; ++k) acc += tb[row * 9 + k] * wih[g * 9 + k];
  xg[(size_t)row * 256 + g] = acc;
}

// ---------------------------------------------------------------------------
// LSTM layer1 input projection: xg.T-tiles = wih @ hs0.T via WMMA.
// grid = 160 M-tiles, block = 128 (4 waves x 4 n-tiles).
// ---------------------------------------------------------------------------
__global__ __launch_bounds__(128) void xgemm1_kernel(
    const float* __restrict__ X, const float* __restrict__ wih,
    float* __restrict__ xg) {
  __shared__ alignas(32) _Float16 sWA[32 * 512];  // 16 nt x 2 kt frags
  __shared__ alignas(32) _Float16 sA[16 * 64];
  const int tid = threadIdx.x, wave = tid >> 5, lane = tid & 31;
  stage_a_sw<64, 256>(sWA, wih, tid, 128);
  const int m0 = blockIdx.x * 16;
  for (int i = tid * 4; i < 16 * 64; i += 512) {
    const float4 v = *(const float4*)(X + (size_t)m0 * 64 + i);
    v4h o; o[0] = (_Float16)v.x; o[1] = (_Float16)v.y;
           o[2] = (_Float16)v.z; o[3] = (_Float16)v.w;
    *(v4h*)(sA + i) = o;
  }
  __syncthreads();
  v16h bf0 = load_bT(sA, 64, 0, lane);
  v16h bf1 = load_bT(sA, 64, 1, lane);
  for (int nt = wave * 4; nt < wave * 4 + 4; ++nt) {
    v8f c;
    #pragma unroll
    for (int r = 0; r < 8; ++r) c[r] = 0.f;
    c = wmma16(load_frag_sw(sWA + (nt * 2 + 0) * 512, lane), bf0, c);
    c = wmma16(load_frag_sw(sWA + (nt * 2 + 1) * 512, lane), bf1, c);
    store_dT_f32(xg + (size_t)m0 * 256, 256, nt, lane, c);
  }
}

// ---------------------------------------------------------------------------
// LSTM recurrence: g = xg[t] + bias + h @ whh.T (gate order i,f,g,o).
// grid = 16 (16 batch rows), block = 32 (one wave). Ping-pong h in LDS.
// ---------------------------------------------------------------------------
__global__ __launch_bounds__(32) void lstm_step_kernel(
    const float* __restrict__ xg,   // [256][T][256]
    const float* __restrict__ whh,  // [256][64]
    const float* __restrict__ bih, const float* __restrict__ bhh,
    float* __restrict__ hs_out,     // layer0: [256][T][64] or nullptr
    float* __restrict__ hlast_out,  // layer1: h-matrix base or nullptr
    int T) {
  __shared__ alignas(32) _Float16 sWA[32 * 512];  // whh: 16 nt x 2 kt frags
  __shared__ alignas(32) _Float16 sH[2][16 * 64]; // [b][u] f16
  __shared__ alignas(32) float    sC[16 * 64];
  __shared__ alignas(32) float    sBias[256];
  const int lane = threadIdx.x;
  const int b0 = blockIdx.x * 16;
  stage_a_sw<64, 256>(sWA, whh, lane, 32);
  for (int i = lane; i < 256; i += 32) sBias[i] = bih[i] + bhh[i];
  for (int i = lane * 8; i < 16 * 64; i += 256) {
    v8h z;
    #pragma unroll
    for (int q = 0; q < 8; ++q) z[q] = (_Float16)0.f;
    *(v8h*)(sH[0] + i) = z;
    *(float4*)(sC + i) = make_float4(0.f, 0.f, 0.f, 0.f);
    *(float4*)(sC + i + 4) = make_float4(0.f, 0.f, 0.f, 0.f);
  }
  __syncthreads();

  const int brow = lane & 15, hb = ((lane >> 4) << 3);
  for (int t = 0; t < T; ++t) {
    const _Float16* Hc = sH[t & 1];
    _Float16*       Hn = sH[(t + 1) & 1];
    const v16h bf0 = load_bT(Hc, 64, 0, lane);
    const v16h bf1 = load_bT(Hc, 64, 1, lane);
    for (int ug = 0; ug < 4; ++ug) {
      v8f cg[4];
      #pragma unroll
      for (int gt = 0; gt < 4; ++gt) {  // i, f, g, o
        const int nt = gt * 4 + ug;     // gate tile: g0 = nt*16
        v8f c = bias_rows(sBias, nt, lane);
        c = wmma16(load_frag_sw(sWA + (nt * 2 + 0) * 512, lane), bf0, c);
        c = wmma16(load_frag_sw(sWA + (nt * 2 + 1) * 512, lane), bf1, c);
        const float* xp = xg + ((size_t)(b0 + brow) * T + t) * 256 + nt * 16 + hb;
        const float4 xa = *(const float4*)xp;
        const float4 xb = *(const float4*)(xp + 4);
        c[0] += xa.x; c[1] += xa.y; c[2] += xa.z; c[3] += xa.w;
        c[4] += xb.x; c[5] += xb.y; c[6] += xb.z; c[7] += xb.w;
        cg[gt] = c;
      }
      const int u0 = ug * 16 + hb;      // 8 contiguous hidden units
      float* cp = sC + brow * 64 + u0;
      const float4 c0 = *(const float4*)cp;
      const float4 c1 = *(const float4*)(cp + 4);
      const float cold[8] = {c0.x, c0.y, c0.z, c0.w, c1.x, c1.y, c1.z, c1.w};
      float cn[8], hn[8];
      v8h hv;
      #pragma unroll
      for (int r = 0; r < 8; ++r) {
        const float iv = sigmoidf_(cg[0][r]);
        const float fv = sigmoidf_(cg[1][r]);
        const float gv = tanhf(cg[2][r]);
        const float ov = sigmoidf_(cg[3][r]);
        cn[r] = fv * cold[r] + iv * gv;
        hn[r] = ov * tanhf(cn[r]);
        hv[r] = (_Float16)hn[r];
      }
      *(float4*)cp       = make_float4(cn[0], cn[1], cn[2], cn[3]);
      *(float4*)(cp + 4) = make_float4(cn[4], cn[5], cn[6], cn[7]);
      *(v8h*)(Hn + brow * 64 + u0) = hv;
      if (hs_out) {
        float* hp = hs_out + ((size_t)(b0 + brow) * T + t) * 64 + u0;
        *(float4*)hp       = make_float4(hn[0], hn[1], hn[2], hn[3]);
        *(float4*)(hp + 4) = make_float4(hn[4], hn[5], hn[6], hn[7]);
      }
      if (hlast_out && t == T - 1) {
        float* hp = hlast_out + (size_t)(256 + b0 + brow) * 128 + 64 + u0;
        *(float4*)hp       = make_float4(hn[0], hn[1], hn[2], hn[3]);
        *(float4*)(hp + 4) = make_float4(hn[4], hn[5], hn[6], hn[7]);
      }
    }
  }
}

// ---------------------------------------------------------------------------
// EdgeConv GEMM: out[512,128] = A[512,128] @ W.T (+bias), computed transposed.
// grid = 32 M-tiles, block = 128 (4 waves x 2 n-tiles).
// ---------------------------------------------------------------------------
__global__ __launch_bounds__(128) void gemm128_kernel(
    const float* __restrict__ A, const float* __restrict__ W,
    const float* __restrict__ bias, float* __restrict__ out) {
  __shared__ alignas(32) _Float16 sWA[32 * 512];  // 8 nt x 4 kt frags
  __shared__ alignas(32) _Float16 sA[16 * 128];
  const int tid = threadIdx.x, wave = tid >> 5, lane = tid & 31;
  stage_a_sw<128, 128>(sWA, W, tid, 128);
  const int m0 = blockIdx.x * 16;
  for (int i = tid * 4; i < 16 * 128; i += 512) {
    const float4 v = *(const float4*)(A + (size_t)m0 * 128 + i);
    v4h o; o[0] = (_Float16)v.x; o[1] = (_Float16)v.y;
           o[2] = (_Float16)v.z; o[3] = (_Float16)v.w;
    *(v4h*)(sA + i) = o;
  }
  __syncthreads();
  v16h bf[4];
  #pragma unroll
  for (int ks = 0; ks < 4; ++ks) bf[ks] = load_bT(sA, 128, ks, lane);
  for (int nt = wave * 2; nt < wave * 2 + 2; ++nt) {
    v8f c;
    if (bias) {
      c = bias_rows(bias, nt, lane);
    } else {
      #pragma unroll
      for (int r = 0; r < 8; ++r) c[r] = 0.f;
    }
    #pragma unroll
    for (int ks = 0; ks < 4; ++ks)
      c = wmma16(load_frag_sw(sWA + (nt * 4 + ks) * 512, lane), bf[ks], c);
    store_dT_f32(out + (size_t)m0 * 128, 128, nt, lane, c);
  }
}

// ---------------------------------------------------------------------------
// EdgeConv aggregate: h[i] = relu(max_{j:adj[j,i]!=0 or j==i} Th[j] - Th[i] + bt + Ph[i]).
// grid = 512 vertices, block = 128 features.
// ---------------------------------------------------------------------------
__global__ __launch_bounds__(128) void aggregate_kernel(
    const float* __restrict__ Th, const float* __restrict__ Ph,
    const int* __restrict__ adj, const float* __restrict__ bt,
    float* __restrict__ h) {
  const int i = blockIdx.x, f = threadIdx.x;
  const int V = 512;
  float m = -__builtin_inff();
  for (int j = 0; j < V; ++j) {
    const bool nb = (adj[(size_t)j * V + i] != 0) || (j == i);
    if (nb) m = fmaxf(m, Th[(size_t)j * 128 + f]);
  }
  const float v = m - Th[(size_t)i * 128 + f] + bt[f] + Ph[(size_t)i * 128 + f];
  h[(size_t)i * 128 + f] = fmaxf(v, 0.f);
}

// ---------------------------------------------------------------------------
// Affinity: aff[i,j] = sigmoid(relu((h[j]-h[i]) @ w1.T + b1) @ w2.T + b2).
// grid = 512 (128 edges), block = 256 (8 waves x 16 edges).
// ---------------------------------------------------------------------------
__global__ __launch_bounds__(256) void affinity_kernel(
    const float* __restrict__ h,
    const float* __restrict__ w1, const float* __restrict__ b1,
    const float* __restrict__ w2, const float* __restrict__ b2,
    float* __restrict__ out) {
  __shared__ alignas(32) _Float16 sWA[16 * 512];   // er_w1: 4 nt x 4 kt frags
  __shared__ alignas(32) _Float16 sE[8][16 * 128]; // [edge][k]
  const int tid = threadIdx.x, wave = tid >> 5, lane = tid & 31;
  stage_a_sw<128, 64>(sWA, w1, tid, 256);
  __syncthreads();

  const int e0 = (blockIdx.x * 8 + wave) * 16;  // 16 consecutive edges, same i
  const int irow = e0 >> 8, j0 = e0 & 255;
  _Float16* E = sE[wave];
  for (int x = lane * 4; x < 16 * 128; x += 128) {
    const int p = x >> 7, f = x & 127;
    const float4 hj = *(const float4*)(h + (size_t)(j0 + p) * 128 + f);
    const float4 hi = *(const float4*)(h + (size_t)irow * 128 + f);
    v4h o; o[0] = (_Float16)(hj.x - hi.x); o[1] = (_Float16)(hj.y - hi.y);
           o[2] = (_Float16)(hj.z - hi.z); o[3] = (_Float16)(hj.w - hi.w);
    *(v4h*)(E + x) = o;
  }

  v16h bf[4];
  #pragma unroll
  for (int ks = 0; ks < 4; ++ks) bf[ks] = load_bT(E, 128, ks, lane);

  float part = 0.f;
  #pragma unroll
  for (int nt = 0; nt < 4; ++nt) {
    v8f c = bias_rows(b1, nt, lane);
    #pragma unroll
    for (int ks = 0; ks < 4; ++ks)
      c = wmma16(load_frag_sw(sWA + (nt * 4 + ks) * 512, lane), bf[ks], c);
    const float* wp = w2 + nt * 16 + ((lane >> 4) << 3);
    const float4 wa = *(const float4*)wp;
    const float4 wb = *(const float4*)(wp + 4);
    part += fmaxf(c[0], 0.f) * wa.x + fmaxf(c[1], 0.f) * wa.y +
            fmaxf(c[2], 0.f) * wa.z + fmaxf(c[3], 0.f) * wa.w +
            fmaxf(c[4], 0.f) * wb.x + fmaxf(c[5], 0.f) * wb.y +
            fmaxf(c[6], 0.f) * wb.z + fmaxf(c[7], 0.f) * wb.w;
  }
  part += __shfl_xor(part, 16);  // combine the two feature halves per edge
  if (lane < 16) out[e0 + lane] = sigmoidf_(part + b2[0]);
}

// ---------------------------------------------------------------------------
extern "C" void kernel_launch(void* const* d_in, const int* in_sizes, int n_in,
                              void* d_out, int out_size, void* d_ws, size_t ws_size,
                              hipStream_t stream) {
  (void)in_sizes; (void)n_in; (void)out_size; (void)ws_size;
  const float* det_pts   = (const float*)d_in[0];
  const float* det_boxes = (const float*)d_in[1];
  const float* trk_pts   = (const float*)d_in[2];
  const float* trk_boxes = (const float*)d_in[3];
  const int*   adj       = (const int*)d_in[4];
  const float* pn_w1 = (const float*)d_in[5];  const float* pn_b1 = (const float*)d_in[6];
  const float* pn_w2 = (const float*)d_in[7];  const float* pn_b2 = (const float*)d_in[8];
  const float* pn_w3 = (const float*)d_in[9];  const float* pn_b3 = (const float*)d_in[10];
  const float* dm_w1 = (const float*)d_in[11]; const float* dm_b1 = (const float*)d_in[12];
  const float* dm_w2 = (const float*)d_in[13]; const float* dm_b2 = (const float*)d_in[14];
  const float* l0_wih = (const float*)d_in[15]; const float* l0_whh = (const float*)d_in[16];
  const float* l0_bih = (const float*)d_in[17]; const float* l0_bhh = (const float*)d_in[18];
  const float* l1_wih = (const float*)d_in[19]; const float* l1_whh = (const float*)d_in[20];
  const float* l1_bih = (const float*)d_in[21]; const float* l1_bhh = (const float*)d_in[22];
  const float* gc_wt = (const float*)d_in[23]; const float* gc_bt = (const float*)d_in[24];
  const float* gc_wp = (const float*)d_in[25]; const float* gc_bp = (const float*)d_in[26];
  const float* er_w1 = (const float*)d_in[27]; const float* er_b1 = (const float*)d_in[28];
  const float* er_w2 = (const float*)d_in[29]; const float* er_b2 = (const float*)d_in[30];

  float* ws  = (float*)d_ws;
  float* h   = ws;                       // [512][128]
  float* hs0 = ws + 65536;               // [256][10][64]
  float* xg  = hs0 + 163840;             // [256][10][256]
  float* Th  = xg + 655360;              // [512][128]
  float* Ph  = Th + 65536;               // [512][128]
  float* out = (float*)d_out;            // [2][256][256]

  pointnet_kernel<<<512, 128, 0, stream>>>(det_pts, trk_pts,
      pn_w1, pn_b1, pn_w2, pn_b2, pn_w3, pn_b3, h);
  detmot_kernel<<<256, 64, 0, stream>>>(det_boxes, dm_w1, dm_b1, dm_w2, dm_b2, h);
  xgemm0_kernel<<<2560, 256, 0, stream>>>(trk_boxes, l0_wih, xg);
  lstm_step_kernel<<<16, 32, 0, stream>>>(xg, l0_whh, l0_bih, l0_bhh, hs0, nullptr, 10);
  xgemm1_kernel<<<160, 128, 0, stream>>>(hs0, l1_wih, xg);
  lstm_step_kernel<<<16, 32, 0, stream>>>(xg, l1_whh, l1_bih, l1_bhh, nullptr, h, 10);
  for (int k = 0; k < 4; ++k) {
    gemm128_kernel<<<32, 128, 0, stream>>>(h, gc_wt + (size_t)k * 128 * 128, nullptr, Th);
    gemm128_kernel<<<32, 128, 0, stream>>>(h, gc_wp + (size_t)k * 128 * 128, gc_bp + k * 128, Ph);
    aggregate_kernel<<<512, 128, 0, stream>>>(Th, Ph, adj, gc_bt + k * 128, h);
    if (k == 0)
      affinity_kernel<<<512, 256, 0, stream>>>(h, er_w1, er_b1, er_w2, er_b2, out);
  }
  affinity_kernel<<<512, 256, 0, stream>>>(h, er_w1, er_b1, er_w2, er_b2, out + 65536);
}
